// DeformableAttention_45337674776967
// MI455X (gfx1250) — matrix operand order repaired and checked by reading, not compiled, not run on hardware
//
#include <hip/hip_runtime.h>
#include <hip/hip_bf16.h>

// ---------------------------------------------------------------------------
// Problem constants (from the reference)
// ---------------------------------------------------------------------------
#define B_    4
#define Q_    8192
#define E_    256
#define H_    8
#define L_    4
#define P_    4
#define HD_   32
#define VLEN_ 21760   // 128*128 + 64*64 + 32*32 + 16*16

typedef __bf16  v16bf __attribute__((ext_vector_type(16)));
typedef __bf16  v8bf  __attribute__((ext_vector_type(8)));
typedef float   v8f   __attribute__((ext_vector_type(8)));

__device__ __forceinline__ __bf16 f32_to_bf16(float f) {
    unsigned u = __builtin_bit_cast(unsigned, f);
    u += 0x7FFFu + ((u >> 16) & 1u);        // round-to-nearest-even
    unsigned short h = (unsigned short)(u >> 16);
    return __builtin_bit_cast(__bf16, h);
}
__device__ __forceinline__ float bf16_to_f32(__bf16 b) {
    unsigned short h = __builtin_bit_cast(unsigned short, b);
    unsigned u = ((unsigned)h) << 16;
    return __builtin_bit_cast(float, u);
}

// ---------------------------------------------------------------------------
// One-shot fp32 -> bf16 conversion (memory bound, 8 elements/thread).
// n must be a multiple of 8 (true for all uses).
// ---------------------------------------------------------------------------
__global__ void cvt_f32_bf16_kernel(const float* __restrict__ src,
                                    __bf16* __restrict__ dst, long n)
{
    long i8 = ((long)blockIdx.x * blockDim.x + threadIdx.x) * 8;
    if (i8 >= n) return;
    float4 a = *(const float4*)(src + i8);
    float4 b = *(const float4*)(src + i8 + 4);
    v8bf o;
    o[0] = f32_to_bf16(a.x); o[1] = f32_to_bf16(a.y);
    o[2] = f32_to_bf16(a.z); o[3] = f32_to_bf16(a.w);
    o[4] = f32_to_bf16(b.x); o[5] = f32_to_bf16(b.y);
    o[6] = f32_to_bf16(b.z); o[7] = f32_to_bf16(b.w);
    *(v8bf*)(dst + i8) = o;
}

// ---------------------------------------------------------------------------
// WMMA GEMM:  C[M,N] = act( A[M,K] @ W[N,K]^T + bias[N] ),  A/W bf16, acc f32.
// Block: 128 threads = 4 waves; block tile 64x64; wave tile 16x64.
// Weight panel (64 cols x K) staged in LDS (32 KB), B fragments via
// ds_load_b128; A fragments via global_load_b128.  K compile-time (=256)
// so the K loop fully unrolls.  ACT: 0=none, 1=tanh.  OUTBF: C dtype.
// Requires M%64==0, N%64==0.
// ---------------------------------------------------------------------------
template <int ACT, int OUTBF, int K>
__global__ __launch_bounds__(128)
void gemm_bf16_wmma(const __bf16* __restrict__ A,
                    const __bf16* __restrict__ W,
                    const float* __restrict__ bias,
                    void* __restrict__ Cv,
                    int M, int N)
{
    __shared__ __bf16 ldsW[64 * K];                 // 32 KB for K=256

    const int lane = threadIdx.x & 31;
    const int wave = threadIdx.x >> 5;              // 0..3
    const int l16  = lane & 15;
    const int hi   = lane >> 4;                     // lane half

    const int m0 = blockIdx.y * 64 + wave * 16;
    const int n0 = blockIdx.x * 64;

    // ---- stage the 64-column weight panel into LDS (rows n0..n0+63) ----
    {
        const uint4* src = (const uint4*)(W + (size_t)n0 * K);
        uint4* dst = (uint4*)ldsW;
        #pragma unroll
        for (int i = threadIdx.x; i < (64 * K) / 8; i += 128)
            dst[i] = src[i];
    }
    __syncthreads();

    const __bf16* arow = A + (size_t)(m0 + l16) * K;
    const int ah = hi * 16;

    v8f acc[4] = {};

    #pragma unroll
    for (int k0 = 0; k0 < K; k0 += 32) {
        // A fragment (16x32): lane half 0 -> K k0..k0+15,
        //                     lane half 1 -> K k0+16..k0+31
        v8bf alo = *(const v8bf*)(arow + k0 + ah);
        v8bf ahiv = *(const v8bf*)(arow + k0 + 8 + ah);
        v16bf af = __builtin_shufflevector(alo, ahiv,
                       0,1,2,3,4,5,6,7,8,9,10,11,12,13,14,15);
        if (k0 + 32 < K)
            __builtin_prefetch(arow + k0 + 32, 0, 1);   // global_prefetch_b8

        // B fragments (32x16 each): lane half 0 -> K k0..k0+15,
        // half 1 -> k0+16..k0+31; column = lane&15.  B[k][n] == W[n][k].
        const int kb = k0 + (hi << 4);
        #pragma unroll
        for (int s = 0; s < 4; ++s) {
            const v8bf* wp = (const v8bf*)&ldsW[(s * 16 + l16) * K + kb];
            v8bf b0 = wp[0];
            v8bf b1 = wp[1];
            v16bf bf = __builtin_shufflevector(b0, b1,
                           0,1,2,3,4,5,6,7,8,9,10,11,12,13,14,15);
            acc[s] = __builtin_amdgcn_wmma_f32_16x16x32_bf16(
                         false, af, false, bf, (short)0, acc[s], false, false);
        }
    }

    // ---- epilogue: VGPR r holds row (r + 8*hi), col = lane&15 ----
    #pragma unroll
    for (int s = 0; s < 4; ++s) {
        const int col = n0 + s * 16 + l16;
        const float bv = bias ? bias[col] : 0.0f;
        #pragma unroll
        for (int r = 0; r < 8; ++r) {
            const int row = m0 + r + hi * 8;
            float v = acc[s][r] + bv;
            if (ACT == 1) v = tanhf(v);
            if (OUTBF) ((__bf16*)Cv)[(size_t)row * N + col] = f32_to_bf16(v);
            else       ((float*)Cv)[(size_t)row * N + col] = v;
        }
    }
}

// ---------------------------------------------------------------------------
// Softmax over contiguous groups of 16 (one thread per (b,q,h) group).
// ---------------------------------------------------------------------------
__global__ void softmax16_kernel(float* __restrict__ x, int ngroups)
{
    int g = blockIdx.x * blockDim.x + threadIdx.x;
    if (g >= ngroups) return;
    float* p = x + (size_t)g * 16;
    float m = -3.402823466e38f;
    float v[16];
    #pragma unroll
    for (int i = 0; i < 16; ++i) { v[i] = p[i]; m = fmaxf(m, v[i]); }
    float s = 0.f;
    #pragma unroll
    for (int i = 0; i < 16; ++i) { v[i] = __expf(v[i] - m); s += v[i]; }
    float inv = 1.0f / s;
    #pragma unroll
    for (int i = 0; i < 16; ++i) p[i] = v[i] * inv;
}

// ---------------------------------------------------------------------------
// Deformable sampling + weighted accumulation (bf16 value, bf16 out).
// Grid: B*Q blocks of 256 threads; wave per head; lane = channel.
// Corner fetches: 32 contiguous bf16 (64B/wave), L2-resident (v = 44.6MB).
// ---------------------------------------------------------------------------
__global__ __launch_bounds__(256)
void msda_sample_kernel(const float* __restrict__ ref,      // (B,Q,L,2)
                        const float* __restrict__ off,      // (B,Q,H,L,P,2)
                        const float* __restrict__ attn,     // (B,Q,H,L*P)
                        const __bf16* __restrict__ v,       // (B,VLEN,E) bf16
                        __bf16* __restrict__ out_acc)       // (B,Q,E) bf16
{
    const int bq   = blockIdx.x;
    const int b    = bq / Q_;
    const int h    = threadIdx.x >> 5;
    const int lane = threadIdx.x & 31;

    const float* refp = ref  + (size_t)bq * (L_ * 2);
    const float* offp = off  + ((size_t)bq * H_ + h) * (L_ * P_ * 2);
    const float* ap   = attn + ((size_t)bq * H_ + h) * (L_ * P_);

    const int dims[4]  = {128, 64, 32, 16};
    const int start[4] = {0, 16384, 20480, 21504};

    float acc = 0.0f;

    #pragma unroll
    for (int l = 0; l < L_; ++l) {
        const float rx = refp[l * 2 + 0];
        const float ry = refp[l * 2 + 1];
        const int   Wl = dims[l];
        const float sc = 0.5f * (float)(Wl - 1);
        const __bf16* vb = v + ((size_t)b * VLEN_ + start[l]) * E_ + h * HD_ + lane;

        #pragma unroll
        for (int p = 0; p < P_; ++p) {
            const int s = l * P_ + p;
            float lx = rx + offp[s * 2 + 0];
            float ly = ry + offp[s * 2 + 1];
            lx = fminf(fmaxf(lx, -1.0f), 1.0f);
            ly = fminf(fmaxf(ly, -1.0f), 1.0f);
            const float a = ap[s];

            const float x  = (lx + 1.0f) * sc;
            const float y  = (ly + 1.0f) * sc;
            const float xf = floorf(x);
            const float yf = floorf(y);
            const float wx = x - xf;
            const float wy = y - yf;

            int x0 = (int)xf; x0 = min(max(x0, 0), Wl - 1);
            int x1 = min(x0 + 1, Wl - 1);
            int y0 = (int)yf; y0 = min(max(y0, 0), Wl - 1);
            int y1 = min(y0 + 1, Wl - 1);

            const float v00 = bf16_to_f32(vb[(size_t)(y0 * Wl + x0) * E_]);
            const float v01 = bf16_to_f32(vb[(size_t)(y0 * Wl + x1) * E_]);
            const float v10 = bf16_to_f32(vb[(size_t)(y1 * Wl + x0) * E_]);
            const float v11 = bf16_to_f32(vb[(size_t)(y1 * Wl + x1) * E_]);

            const float top = v00 + wx * (v01 - v00);
            const float bot = v10 + wx * (v11 - v10);
            acc += a * (top + wy * (bot - top));
        }
    }

    out_acc[(size_t)bq * E_ + h * HD_ + lane] = f32_to_bf16(acc);
}

// ---------------------------------------------------------------------------
// Launch.  Workspace layout (bf16 unless noted), ~173 MB total:
//   value_bf   22,282,240   queries_bf 8,388,608
//   wV 65,536  wOff 65,536  wAttn 32,768  wOut 65,536
//   v_proj_bf  22,282,240
//   off  (f32) 8,388,608    attn (f32) 4,194,304
//   out_acc_bf 8,388,608
// ---------------------------------------------------------------------------
extern "C" void kernel_launch(void* const* d_in, const int* in_sizes, int n_in,
                              void* d_out, int out_size, void* d_ws, size_t ws_size,
                              hipStream_t stream)
{
    const float* queries = (const float*)d_in[0];
    const float* ref     = (const float*)d_in[1];
    const float* value   = (const float*)d_in[2];
    const float* V_W     = (const float*)d_in[4];
    const float* off_W   = (const float*)d_in[5];
    const float* off_b   = (const float*)d_in[6];
    const float* attn_W  = (const float*)d_in[7];
    const float* attn_b  = (const float*)d_in[8];
    const float* out_W   = (const float*)d_in[9];
    float*       out     = (float*)d_out;

    const size_t NV = (size_t)B_ * VLEN_ * E_;   // 22,282,240
    const size_t NQ = (size_t)B_ * Q_ * E_;      //  8,388,608

    __bf16* value_bf   = (__bf16*)d_ws;
    __bf16* queries_bf = value_bf   + NV;
    __bf16* wV         = queries_bf + NQ;
    __bf16* wOff       = wV    + 256 * 256;
    __bf16* wAttn      = wOff  + 256 * 256;
    __bf16* wOut       = wAttn + 128 * 256;
    __bf16* v_proj     = wOut  + 256 * 256;
    float*  off_buf    = (float*)(v_proj + NV);
    float*  attn_buf   = off_buf + (size_t)(B_ * Q_) * 256;
    __bf16* out_acc    = (__bf16*)(attn_buf + (size_t)(B_ * Q_) * 128);

    const int MQ = B_ * Q_;        // 32768
    const int MV = B_ * VLEN_;     // 87040
    dim3 blk(128);

    // 0) one-shot fp32 -> bf16 conversions (memory bound)
    auto cvt = [&](const float* s, __bf16* d, long n) {
        cvt_f32_bf16_kernel<<<(unsigned)((n / 8 + 255) / 256), 256, 0, stream>>>(s, d, n);
    };
    cvt(value,   value_bf,   (long)NV);
    cvt(queries, queries_bf, (long)NQ);
    cvt(V_W,     wV,    256 * 256);
    cvt(off_W,   wOff,  256 * 256);
    cvt(attn_W,  wAttn, 128 * 256);
    cvt(out_W,   wOut,  256 * 256);

    // 1) v_proj = value @ V_W^T  (bf16 out)          87040 x 256 x 256
    gemm_bf16_wmma<0, 1, 256><<<dim3(4, MV / 64), blk, 0, stream>>>(
        value_bf, wV, nullptr, (void*)v_proj, MV, 256);

    // 2) off = tanh(queries @ off_W^T + off_b)       32768 x 256 x 256
    gemm_bf16_wmma<1, 0, 256><<<dim3(4, MQ / 64), blk, 0, stream>>>(
        queries_bf, wOff, off_b, (void*)off_buf, MQ, 256);

    // 3) attn logits = queries @ attn_W^T + attn_b   32768 x 128 x 256
    gemm_bf16_wmma<0, 0, 256><<<dim3(2, MQ / 64), blk, 0, stream>>>(
        queries_bf, wAttn, attn_b, (void*)attn_buf, MQ, 128);

    // 4) softmax over each (b,q,h) group of 16
    {
        int ngroups = MQ * H_;
        softmax16_kernel<<<(ngroups + 255) / 256, 256, 0, stream>>>(attn_buf, ngroups);
    }

    // 5) deformable bilinear sampling + accumulation (bf16 in/out)
    msda_sample_kernel<<<MQ, 256, 0, stream>>>(ref, off_buf, attn_buf, v_proj, out_acc);

    // 6) out = out_acc @ out_W^T  (f32 out)          32768 x 256 x 256
    gemm_bf16_wmma<0, 0, 256><<<dim3(4, MQ / 64), blk, 0, stream>>>(
        out_acc, wOut, nullptr, (void*)out, MQ, 256);
}
